// MultiHeadAttentionModule_37812892074469
// MI455X (gfx1250) — matrix operand order repaired and checked by reading, not compiled, not run on hardware
//
#include <hip/hip_runtime.h>
#include <hip/hip_bf16.h>

// ---------------------------------------------------------------------------
// MHA for MI455X (gfx1250, wave32, WMMA). bf16 matrix math, f32 accumulate.
// Round 3: async global->LDS staging with correctly-typed builtin args.
// ---------------------------------------------------------------------------

typedef __attribute__((ext_vector_type(16))) __bf16 v16bf;
typedef __attribute__((ext_vector_type(8)))  __bf16 v8bf;
typedef __attribute__((ext_vector_type(4)))  __bf16 v4bf;
typedef __attribute__((ext_vector_type(8)))  float  v8f;
typedef __attribute__((ext_vector_type(4)))  int    v4i;

#define DIM_C 1024
#define N_SEQ 2048
#define N_B   2
#define N_H   16
#define D_H   64

// --- Async global->LDS copy (gfx1250 GLOBAL_LOAD_ASYNC_TO_LDS_B128) ---------
#if defined(__gfx1250__) && __has_builtin(__builtin_amdgcn_global_load_async_to_lds_b128)
#define USE_ASYNC_LDS 1
#else
#define USE_ASYNC_LDS 0
#endif

#if USE_ASYNC_LDS
typedef __attribute__((address_space(1))) v4i g_v4i;   // global (prints as __device__)
typedef __attribute__((address_space(3))) v4i l_v4i;   // LDS    (prints as __shared__)
__device__ __forceinline__ void async_cp16(const void* g, void* l) {
  __builtin_amdgcn_global_load_async_to_lds_b128((g_v4i*)g, (l_v4i*)l,
                                                 /*offset=*/0, /*cpol=*/0);
}
__device__ __forceinline__ void wait_async0() {
#if __has_builtin(__builtin_amdgcn_s_wait_asynccnt)
  __builtin_amdgcn_s_wait_asynccnt(0);
#else
  asm volatile("s_wait_asynccnt 0" ::: "memory");
#endif
}
#endif

union Frag16 { v16bf v; v8bf h[2]; };

// A-fragment (16x32, MxK): lane m (0..15), half hh: elements e=0..7 -> K=8*hh+e,
// e=8..15 -> K=16+8*hh+(e-8). Two contiguous 16B loads.
__device__ __forceinline__ v16bf load_frag_a(const __bf16* base, int stride) {
  const int lane = threadIdx.x & 31;
  const int m = lane & 15, hh = lane >> 4;
  const __bf16* p = base + m * stride + 8 * hh;
  Frag16 f;
  f.h[0] = *(const v8bf*)(p);
  f.h[1] = *(const v8bf*)(p + 16);
  return f.v;
}

// B-fragment (32x16, KxN): lane n (0..15), half hh: elements e=0..7 -> K=8*hh+e,
// e=8..15 -> K=16+8*hh+(e-8).
// Source stored as [N][K] row-major (i.e. each lane reads two 16B runs, 32 bytes apart).
__device__ __forceinline__ v16bf load_frag_b(const __bf16* base, int stride) {
  const int lane = threadIdx.x & 31;
  const int n = lane & 15, hh = lane >> 4;
  const __bf16* p = base + n * stride + 8 * hh;
  Frag16 f;
  f.h[0] = *(const v8bf*)(p);
  f.h[1] = *(const v8bf*)(p + 16);
  return f.v;
}

__device__ __forceinline__ v8f wmma_bf16(v16bf a, v16bf b, v8f c) {
  // D = A(16x32 bf16) x B(32x16 bf16) + C(16x16 f32)
  return __builtin_amdgcn_wmma_f32_16x16x32_bf16(
      /*neg_a=*/false, a, /*neg_b=*/false, b,
      /*c_mod=*/(short)0, c, /*reuse_a=*/false, /*reuse_b=*/false);
}

// ---------------------------------------------------------------------------
// f32 -> bf16 conversion (vectorized, 4 elems/thread)
// ---------------------------------------------------------------------------
__global__ void __launch_bounds__(256)
cvt_f32_to_bf16(const float* __restrict__ in, __bf16* __restrict__ out, int nvec) {
  int i = blockIdx.x * blockDim.x + threadIdx.x;
  if (i < nvec) {
    float4 f = ((const float4*)in)[i];
    v4bf o;
    o[0] = (__bf16)f.x; o[1] = (__bf16)f.y; o[2] = (__bf16)f.z; o[3] = (__bf16)f.w;
    *(v4bf*)(out + (size_t)i * 4) = o;
  }
}

// ---------------------------------------------------------------------------
// Tiled WMMA GEMM: Y[M x 1024] = A[M x K] * W[1024 x K]^T
// Block = 256 threads = 8 waves (2 m-waves x 4 n-waves). Block tile 64x128,
// wave tile 32x32 (2x2 WMMA tiles). LDS-staged (async if available), K step 32.
// mode 0: write bf16 [B,H,N,Dh]   (Q, K layout)
// mode 1: write bf16 [B,H,Dh,N]   (V transposed for PV B-fragments)
// mode 2: write f32 row-major + bias (final output projection)
// ---------------------------------------------------------------------------
#define BM 64
#define BN 128
#define BK 32
#define LDT (BK + 8)   // padded LDS row stride (bf16 elems), keeps 16B alignment

__global__ void __launch_bounds__(256)
gemm_wmma(const __bf16* __restrict__ A, const __bf16* __restrict__ W,
          __bf16* __restrict__ outb, float* __restrict__ outf,
          const float* __restrict__ bias, int M, int K, int mode) {
  __shared__ __bf16 As[BM][LDT];
  __shared__ __bf16 Bs[BN][LDT];

  const int t    = threadIdx.x;
  const int lane = t & 31;
  const int wave = t >> 5;
  const int mw = wave >> 2;        // 0..1
  const int nw = wave & 3;         // 0..3
  const int mBase = blockIdx.x * BM;
  const int nBase = blockIdx.y * BN;

  v8f acc[2][2] = {};

  const int arow = t >> 2, acol = (t & 3) * 8;   // 64 rows x 32 cols, 8 elems/thread
  const int brow = t >> 1, bcol = (t & 1) * 16;  // 128 rows x 32 cols, 16 elems/thread

  for (int k0 = 0; k0 < K; k0 += BK) {
#if USE_ASYNC_LDS
    // ASYNCcnt-tracked DMA into LDS: no VGPR round-trip for staging data.
    async_cp16(&A[(size_t)(mBase + arow) * K + k0 + acol], &As[arow][acol]);
    async_cp16(&W[(size_t)(nBase + brow) * K + k0 + bcol], &Bs[brow][bcol]);
    async_cp16(&W[(size_t)(nBase + brow) * K + k0 + bcol + 8], &Bs[brow][bcol + 8]);
    wait_async0();
#else
    *(v8bf*)&As[arow][acol] =
        *(const v8bf*)&A[(size_t)(mBase + arow) * K + k0 + acol];
    *(v8bf*)&Bs[brow][bcol] =
        *(const v8bf*)&W[(size_t)(nBase + brow) * K + k0 + bcol];
    *(v8bf*)&Bs[brow][bcol + 8] =
        *(const v8bf*)&W[(size_t)(nBase + brow) * K + k0 + bcol + 8];
#endif
    __syncthreads();

    v16bf a0 = load_frag_a(&As[mw * 32][0], LDT);
    v16bf a1 = load_frag_a(&As[mw * 32 + 16][0], LDT);
    v16bf b0 = load_frag_b(&Bs[nw * 32][0], LDT);
    v16bf b1 = load_frag_b(&Bs[nw * 32 + 16][0], LDT);

    acc[0][0] = wmma_bf16(a0, b0, acc[0][0]);
    acc[0][1] = wmma_bf16(a0, b1, acc[0][1]);
    acc[1][0] = wmma_bf16(a1, b0, acc[1][0]);
    acc[1][1] = wmma_bf16(a1, b1, acc[1][1]);
    __syncthreads();
  }

  // Epilogue. C/D layout: lane = n + 16*hh, VGPR r -> (M = 8*hh + r, N = n)
  const int n = lane & 15, hh = lane >> 4;
#pragma unroll
  for (int mt = 0; mt < 2; ++mt) {
#pragma unroll
    for (int nt = 0; nt < 2; ++nt) {
#pragma unroll
      for (int r = 0; r < 8; ++r) {
        int row = mBase + mw * 32 + mt * 16 + hh * 8 + r;
        int col = nBase + nw * 32 + nt * 16 + n;
        float v = acc[mt][nt][r];
        if (mode == 2) {
          outf[(size_t)row * DIM_C + col] = v + bias[col];
        } else {
          int b = row >> 11, nq = row & (N_SEQ - 1);
          int h = col >> 6, d = col & (D_H - 1);
          size_t idx = (mode == 0)
              ? ((size_t)(b * N_H + h) * N_SEQ + nq) * D_H + d     // [B,H,N,Dh]
              : ((size_t)(b * N_H + h) * D_H + d) * N_SEQ + nq;    // [B,H,Dh,N]
          outb[idx] = (__bf16)v;
        }
      }
    }
  }
}

// ---------------------------------------------------------------------------
// Flash attention: one wave per 16-row Q tile. 8 waves/block -> 128 q rows.
// KV streamed in chunks of 32. Online softmax with cross-lane (width-16)
// reductions; P bounced through per-wave LDS to re-fragment C->A layout.
// ---------------------------------------------------------------------------
__global__ void __launch_bounds__(256)
attn_flash(const __bf16* __restrict__ Qb, const __bf16* __restrict__ Kb,
           const __bf16* __restrict__ Vtb, const int* __restrict__ mask,
           __bf16* __restrict__ Ob) {
  __shared__ __bf16 Ps[8][16][40];   // per-wave 16x32 P tile (padded rows)

  const int t = threadIdx.x, lane = t & 31, wave = t >> 5;
  const int n = lane & 15, hh = lane >> 4;

  const int qchunks = N_SEQ / 128;              // 16
  const int bh = blockIdx.x / qchunks;          // 0..31  (= b*16 + h)
  const int q0 = (blockIdx.x % qchunks) * 128 + wave * 16;
  const int b = bh >> 4;

  // Q fragments kept resident (A layout), d in [0,32) and [32,64)
  const __bf16* Qrow = Qb + ((size_t)bh * N_SEQ + q0) * D_H;
  v16bf aQ0 = load_frag_a(Qrow, D_H);
  v16bf aQ1 = load_frag_a(Qrow + 32, D_H);

  v8f o0 = {}, o1 = {}, o2 = {}, o3 = {};
  float l[8], mrow[8];
#pragma unroll
  for (int r = 0; r < 8; ++r) { l[r] = 0.f; mrow[r] = -1e30f; }

  const float scale = 0.125f;  // 1/sqrt(64)
  const __bf16* Kbh = Kb + (size_t)bh * N_SEQ * D_H;
  const __bf16* Vbh = Vtb + (size_t)bh * D_H * N_SEQ;
  const int* maskb = mask + b * N_SEQ;

  for (int kv0 = 0; kv0 < N_SEQ; kv0 += 32) {
    // Warm WGP$/L2 for the next chunk (global_prefetch_b8).
    if (kv0 + 32 < N_SEQ) {
      __builtin_prefetch(Kbh + (size_t)(kv0 + 32 + n) * D_H, 0, 3);
      __builtin_prefetch(Kbh + (size_t)(kv0 + 48 + n) * D_H, 0, 3);
      __builtin_prefetch(Vbh + (size_t)lane * N_SEQ + kv0 + 32, 0, 3);
    }

    // ---- S = Q K^T for two 16-col tiles (4 WMMAs) ----
    const __bf16* Krow0 = Kbh + (size_t)kv0 * D_H;
    const __bf16* Krow1 = Kbh + (size_t)(kv0 + 16) * D_H;
    v8f s0 = {}, s1 = {};
    s0 = wmma_bf16(aQ0, load_frag_b(Krow0, D_H), s0);
    s0 = wmma_bf16(aQ1, load_frag_b(Krow0 + 32, D_H), s0);
    s1 = wmma_bf16(aQ0, load_frag_b(Krow1, D_H), s1);
    s1 = wmma_bf16(aQ1, load_frag_b(Krow1 + 32, D_H), s1);

    // ---- mask + online softmax ----
    const int mk0 = maskb[kv0 + n];
    const int mk1 = maskb[kv0 + 16 + n];
    float p0[8], p1[8], mnew[8];
#pragma unroll
    for (int r = 0; r < 8; ++r) {
      float v0 = mk0 ? s0[r] * scale : -1e9f;
      float v1 = mk1 ? s1[r] * scale : -1e9f;
      p0[r] = v0; p1[r] = v1;
      float mx = fmaxf(v0, v1);
      mx = fmaxf(mx, __shfl_xor(mx, 1, 32));
      mx = fmaxf(mx, __shfl_xor(mx, 2, 32));
      mx = fmaxf(mx, __shfl_xor(mx, 4, 32));
      mx = fmaxf(mx, __shfl_xor(mx, 8, 32));
      mnew[r] = fmaxf(mrow[r], mx);
    }
#pragma unroll
    for (int r = 0; r < 8; ++r) {
      float sf = __expf(mrow[r] - mnew[r]);
      mrow[r] = mnew[r];
      float e0 = __expf(p0[r] - mnew[r]);
      float e1 = __expf(p1[r] - mnew[r]);
      p0[r] = e0; p1[r] = e1;
      float sum = e0 + e1;
      sum += __shfl_xor(sum, 1, 32);
      sum += __shfl_xor(sum, 2, 32);
      sum += __shfl_xor(sum, 4, 32);
      sum += __shfl_xor(sum, 8, 32);
      l[r] = l[r] * sf + sum;
      o0[r] *= sf; o1[r] *= sf; o2[r] *= sf; o3[r] *= sf;
    }

    // ---- C-layout P -> LDS -> A-fragment ----
#pragma unroll
    for (int r = 0; r < 8; ++r) {
      Ps[wave][hh * 8 + r][n]      = (__bf16)p0[r];
      Ps[wave][hh * 8 + r][16 + n] = (__bf16)p1[r];
    }
    asm volatile("s_wait_dscnt 0" ::: "memory");
    __builtin_amdgcn_wave_barrier();
    v16bf aP = load_frag_a(&Ps[wave][0][0], 40);

    // ---- O += P V  (V^T stored [Dh][N]: contiguous per lane) ----
    o0 = wmma_bf16(aP, load_frag_b(Vbh + (size_t)0  * N_SEQ + kv0, N_SEQ), o0);
    o1 = wmma_bf16(aP, load_frag_b(Vbh + (size_t)16 * N_SEQ + kv0, N_SEQ), o1);
    o2 = wmma_bf16(aP, load_frag_b(Vbh + (size_t)32 * N_SEQ + kv0, N_SEQ), o2);
    o3 = wmma_bf16(aP, load_frag_b(Vbh + (size_t)48 * N_SEQ + kv0, N_SEQ), o3);
    __builtin_amdgcn_wave_barrier();
  }

  // ---- normalize + store O as bf16, layout [B,N, H*64+d] for final GEMM ----
  const int h = bh & 15;
#pragma unroll
  for (int r = 0; r < 8; ++r) {
    float inv = 1.f / l[r];
    int row = q0 + hh * 8 + r;
    size_t base = ((size_t)b * N_SEQ + row) * DIM_C + h * D_H;
    Ob[base + 0  + n] = (__bf16)(o0[r] * inv);
    Ob[base + 16 + n] = (__bf16)(o1[r] * inv);
    Ob[base + 32 + n] = (__bf16)(o2[r] * inv);
    Ob[base + 48 + n] = (__bf16)(o3[r] * inv);
  }
}

// ---------------------------------------------------------------------------
// Host launcher
// ---------------------------------------------------------------------------
extern "C" void kernel_launch(void* const* d_in, const int* in_sizes, int n_in,
                              void* d_out, int out_size, void* d_ws, size_t ws_size,
                              hipStream_t stream) {
  const float* feat1 = (const float*)d_in[0];   // [2,2048,1024]
  const float* feat2 = (const float*)d_in[1];   // [2,2048,1024]
  const float* Wq    = (const float*)d_in[2];   // [1024,1024]
  const float* Wk    = (const float*)d_in[3];
  const float* Wv    = (const float*)d_in[4];
  const float* Wo    = (const float*)d_in[5];
  const float* bo    = (const float*)d_in[6];   // [1024]
  const int*   mask  = (const int*)d_in[7];     // [2,2048]
  float* out = (float*)d_out;

  const size_t ACT = (size_t)N_B * N_SEQ * DIM_C;   // 4194304
  const size_t WGT = (size_t)DIM_C * DIM_C;         // 1048576

  __bf16* ws = (__bf16*)d_ws;
  size_t off = 0;
  __bf16* Xq  = ws + off; off += ACT;
  __bf16* Xkv = ws + off; off += ACT;
  __bf16* Wqb = ws + off; off += WGT;
  __bf16* Wkb = ws + off; off += WGT;
  __bf16* Wvb = ws + off; off += WGT;
  __bf16* Wob = ws + off; off += WGT;
  __bf16* Qb  = ws + off; off += ACT;
  __bf16* Kb  = ws + off; off += ACT;
  __bf16* Vtb = ws + off; off += ACT;
  __bf16* Ob  = ws + off; off += ACT;

  // 1) f32 -> bf16 conversions
  {
    int nv = (int)(ACT / 4);
    cvt_f32_to_bf16<<<(nv + 255) / 256, 256, 0, stream>>>(feat1, Xq, nv);
    cvt_f32_to_bf16<<<(nv + 255) / 256, 256, 0, stream>>>(feat2, Xkv, nv);
    int nw = (int)(WGT / 4);
    cvt_f32_to_bf16<<<(nw + 255) / 256, 256, 0, stream>>>(Wq, Wqb, nw);
    cvt_f32_to_bf16<<<(nw + 255) / 256, 256, 0, stream>>>(Wk, Wkb, nw);
    cvt_f32_to_bf16<<<(nw + 255) / 256, 256, 0, stream>>>(Wv, Wvb, nw);
    cvt_f32_to_bf16<<<(nw + 255) / 256, 256, 0, stream>>>(Wo, Wob, nw);
  }

  const int M = N_B * N_SEQ;  // 4096
  dim3 ggrid(M / BM, DIM_C / BN);   // (64, 8)

  // 2) Q/K/V projections (WMMA GEMMs)
  gemm_wmma<<<ggrid, 256, 0, stream>>>(Xq,  Wqb, Qb,  nullptr, nullptr, M, DIM_C, 0);
  gemm_wmma<<<ggrid, 256, 0, stream>>>(Xkv, Wkb, Kb,  nullptr, nullptr, M, DIM_C, 0);
  gemm_wmma<<<ggrid, 256, 0, stream>>>(Xkv, Wvb, Vtb, nullptr, nullptr, M, DIM_C, 1);

  // 3) Flash attention: 512 blocks x 8 waves x 16 q-rows = 2*16*2048
  attn_flash<<<N_B * N_H * (N_SEQ / 128), 256, 0, stream>>>(Qb, Kb, Vtb, mask, Ob);

  // 4) Output projection + bias -> f32 d_out
  gemm_wmma<<<ggrid, 256, 0, stream>>>(Ob, Wob, nullptr, out, bo, M, DIM_C, 2);
}
